// ParallelAttention_18537078849675
// MI455X (gfx1250) — matrix operand order repaired and checked by reading, not compiled, not run on hardware
//
#include <hip/hip_runtime.h>
#include <hip/hip_bf16.h>

typedef __attribute__((ext_vector_type(16))) __bf16 v16bf;
typedef __attribute__((ext_vector_type(8)))  __bf16 v8bf;
typedef __attribute__((ext_vector_type(8)))  float  v8f;

#define DEV __device__ __forceinline__

// ---- CDNA5 async global->LDS copy (ASYNCcnt-tracked), GVS addressing ----
// LDS destination offset comes from the low 32 bits of the flat shared address
// (ISA 10.2: LDS aperture truncates flat addr to addr[31:0]).
DEV void async_load_b128(const void* lds_dst, const void* gbase, uint32_t voff_bytes) {
    uint32_t lds_off = (uint32_t)(uintptr_t)lds_dst;
    uint64_t base = (uint64_t)(uintptr_t)gbase;
    asm volatile("global_load_async_to_lds_b128 %0, %1, %2 offset:0"
                 :: "v"(lds_off), "v"(voff_bytes), "s"(base)
                 : "memory");
}
DEV void wait_async0() {
    asm volatile("s_wait_asynccnt 0" ::: "memory");
}

// ---- WMMA fragment gather helpers (wave32; this file's contiguous-k order, same for A and B) ----
// A matrix 16x32 bf16: lanes 0-15 -> M=lane, K=0..15 contiguous;
//                      lanes 16-31 -> M=lane-16, K=16..31 contiguous.
// Caller passes pointer already offset to (row*stride + kbase + half*16).
DEV v16bf frag_a(const __bf16* p0) {
    v8bf lo = *(const v8bf*)p0;
    v8bf hi = *(const v8bf*)(p0 + 8);
    v16bf r;
#pragma unroll
    for (int i = 0; i < 8; ++i) { r[i] = lo[i]; r[i + 8] = hi[i]; }
    return r;
}
// B matrix 32x16 bf16: lanes 0-15 -> N=lane, K=0..15 contiguous;
//                      lanes 16-31 -> N=lane-16, K=16..31 contiguous.
// Caller passes pointer already offset to (rowN*stride + kbase + half*16).
DEV v16bf frag_b(const __bf16* p0) {
    v8bf lo = *(const v8bf*)p0;
    v8bf hi = *(const v8bf*)(p0 + 8);
    v16bf r;
#pragma unroll
    for (int i = 0; i < 8; ++i) { r[i] = lo[i]; r[i + 8] = hi[i]; }
    return r;
}

DEV float redmax16(float v) {
#pragma unroll
    for (int i = 1; i < 16; i <<= 1) v = fmaxf(v, __shfl_xor(v, i, 16));
    return v;
}
DEV float redsum16(float v) {
#pragma unroll
    for (int i = 1; i < 16; i <<= 1) v += __shfl_xor(v, i, 16);
    return v;
}

// ---- elementwise f32 -> bf16 convert ----
__global__ void f32_to_bf16_kernel(const float* __restrict__ src,
                                   __bf16* __restrict__ dst, long n) {
    long i = (long)blockIdx.x * blockDim.x + threadIdx.x;
    if (i < n) dst[i] = (__bf16)src[i];
}

// ---- generic bf16 GEMM:  C[M,N] (f32) = A[M,K] * B[N,K]^T  (both row-major along K)
// block = 256 threads (8 waves), tile 128x128, K-step 32, double-buffered LDS with
// async global->LDS DMA overlapping WMMA compute (one barrier per K-step).
// wave (wm,wn) in 2x4 grid computes a 64x32 sub-tile = 4x2 WMMA fragments.
__global__ __launch_bounds__(256) void gemm_bf16_kernel(
    const __bf16* __restrict__ A, const __bf16* __restrict__ B,
    float* __restrict__ C, int M, int N, int K) {
    __shared__ __bf16 As[2][128][32];
    __shared__ __bf16 Bs[2][128][32];
    const int tid = threadIdx.x;
    const int lane = tid & 31, w = tid >> 5;
    const int half = lane >> 4, lr = lane & 15;
    const int wm = w >> 2, wn = w & 3;
    const int m0 = blockIdx.y * 128, n0 = blockIdx.x * 128;

    v8f zero = {0.f, 0.f, 0.f, 0.f, 0.f, 0.f, 0.f, 0.f};
    v8f acc[4][2];
#pragma unroll
    for (int i = 0; i < 4; ++i)
#pragma unroll
        for (int j = 0; j < 2; ++j) acc[i][j] = zero;

    // per-thread copy slots: 512 chunks of 8 bf16 per tile, 2 per thread
    int rowA[2], colA[2];
#pragma unroll
    for (int i = 0; i < 2; ++i) {
        int chunk = tid + i * 256;
        rowA[i] = chunk >> 2;
        colA[i] = (chunk & 3) * 8;
    }

    // prologue: async-fill buffer 0 with K-step 0
#pragma unroll
    for (int i = 0; i < 2; ++i) {
        async_load_b128(&As[0][rowA[i]][colA[i]], A,
                        (uint32_t)(((size_t)(m0 + rowA[i]) * K + colA[i]) * 2));
        async_load_b128(&Bs[0][rowA[i]][colA[i]], B,
                        (uint32_t)(((size_t)(n0 + rowA[i]) * K + colA[i]) * 2));
    }

    for (int k0 = 0; k0 < K; k0 += 32) {
        const int cur = (k0 >> 5) & 1;
        wait_async0();      // this wave's portion of tile `cur` has landed
        __syncthreads();    // all portions visible; prior reads of `nxt` retired
        if (k0 + 32 < K) {
            const int nxt = cur ^ 1;
            const int kn = k0 + 32;
#pragma unroll
            for (int i = 0; i < 2; ++i) {
                async_load_b128(&As[nxt][rowA[i]][colA[i]], A,
                                (uint32_t)(((size_t)(m0 + rowA[i]) * K + kn + colA[i]) * 2));
                async_load_b128(&Bs[nxt][rowA[i]][colA[i]], B,
                                (uint32_t)(((size_t)(n0 + rowA[i]) * K + kn + colA[i]) * 2));
            }
            if (kn + 32 < K) {
                // pull the K-step after next toward the caches
                __builtin_prefetch(A + (size_t)(m0 + rowA[0]) * K + kn + 32 + colA[0], 0, 1);
                __builtin_prefetch(B + (size_t)(n0 + rowA[0]) * K + kn + 32 + colA[0], 0, 1);
            }
        }
        // compute on buffer `cur` while `nxt` streams in
        v16bf af[4], bfr[2];
#pragma unroll
        for (int fm = 0; fm < 4; ++fm)
            af[fm] = frag_a(&As[cur][wm * 64 + fm * 16 + lr][half * 16]);
#pragma unroll
        for (int fn = 0; fn < 2; ++fn)
            bfr[fn] = frag_b(&Bs[cur][wn * 32 + fn * 16 + lr][half * 16]);
#pragma unroll
        for (int fm = 0; fm < 4; ++fm)
#pragma unroll
            for (int fn = 0; fn < 2; ++fn)
                acc[fm][fn] = __builtin_amdgcn_wmma_f32_16x16x32_bf16(
                    false, af[fm], false, bfr[fn], (short)0, acc[fm][fn], false, false);
    }
    // C fragment layout: lanes 0-15 -> M=r, N=lane; lanes 16-31 -> M=8+r, N=lane-16
#pragma unroll
    for (int fm = 0; fm < 4; ++fm)
#pragma unroll
        for (int fn = 0; fn < 2; ++fn) {
            int row0 = m0 + wm * 64 + fm * 16 + half * 8;
            int col  = n0 + wn * 32 + fn * 16 + lr;
#pragma unroll
            for (int r = 0; r < 8; ++r)
                C[(size_t)(row0 + r) * N + col] = acc[fm][fn][r];
        }
}

// ---- RoPE + split qkv (f32, rows m=s*B+b, cols o in [0,6144)) into bf16 Q/K/V (b,h,s,d) ----
__global__ void rope_split_kernel(const float* __restrict__ qkvf,
                                  __bf16* __restrict__ Qb, __bf16* __restrict__ Kb,
                                  __bf16* __restrict__ Vb) {
    const long idx = (long)blockIdx.x * blockDim.x + threadIdx.x;
    const long total = (long)4096 * 6144;
    if (idx >= total) return;
    const int o = (int)(idx % 6144);
    const int m = (int)(idx / 6144);
    const int s = m >> 1, b = m & 1;
    const int part = o >> 11;          // 0=q, 1=k, 2=v
    const int ho = o & 2047;
    const int h = ho >> 7, d = ho & 127;
    float v = qkvf[idx];
    float outv;
    if (part == 2) {
        outv = v;
    } else {
        const int i = d & 63;
        // inv_freq = 10000^{-i/64}
        const float inv = __expf(-(float)i * (9.210340371976184f / 64.0f));
        const float ang = (float)s * inv;
        const float c = cosf(ang), sn = sinf(ang);
        const int pd = (d < 64) ? d + 64 : d - 64;
        const float vp = qkvf[(size_t)m * 6144 + part * 2048 + h * 128 + pd];
        outv = (d < 64) ? (v * c - vp * sn) : (v * c + vp * sn);
    }
    const size_t dst = (((size_t)(b * 16 + h) * 2048) + s) * 128 + d;
    if (part == 0) Qb[dst] = (__bf16)outv;
    else if (part == 1) Kb[dst] = (__bf16)outv;
    else Vb[dst] = (__bf16)outv;
}

// ---- causal flash attention: grid (S/128, B*NH), 8 waves, wave owns 16 query rows ----
__global__ __launch_bounds__(256) void flash_kernel(
    const __bf16* __restrict__ Q, const __bf16* __restrict__ Kb,
    const __bf16* __restrict__ Vb, __bf16* __restrict__ attn) {
    constexpr int S = 2048, HD = 128;
    __shared__ __bf16 Ks[128][128];     // [key][d]
    __shared__ __bf16 VTs[128][128];    // [d][key]  (transposed V)
    __shared__ __bf16 Ps[8][16][128];   // per-wave P staging (C-layout -> A-layout)
    const int tid = threadIdx.x, lane = tid & 31, w = tid >> 5;
    const int half = lane >> 4, lr = lane & 15;
    const int q0 = blockIdx.x * 128;
    const int bh = blockIdx.y;
    const int b = bh >> 4, h = bh & 15;
    const float scale = 0.08838834764831845f; // 1/sqrt(128)

    // Q fragments straight from global (A-layout)
    const __bf16* Qbase = Q + ((size_t)bh * S + q0 + w * 16) * HD;
    v16bf qf[4];
#pragma unroll
    for (int kd = 0; kd < 4; ++kd)
        qf[kd] = frag_a(Qbase + (size_t)lr * HD + kd * 32 + half * 16);

    v8f zero = {0.f, 0.f, 0.f, 0.f, 0.f, 0.f, 0.f, 0.f};
    v8f o[8];
    float mrow[8], lrow[8];
#pragma unroll
    for (int i = 0; i < 8; ++i) { o[i] = zero; mrow[i] = -3.0e38f; lrow[i] = 0.f; }

    const int ktEnd = q0 >> 7;
    for (int kt = 0; kt <= ktEnd; ++kt) {
        const int k0 = kt * 128;
        __syncthreads();
        const size_t tileBase = ((size_t)bh * S + k0) * HD;
        const __bf16* Vg = Vb + tileBase;
        // K tile: async global->LDS; V tile: load + transpose via ds stores
#pragma unroll
        for (int i = 0; i < 8; ++i) {
            int chunk = tid + i * 256;           // 2048 chunks of 8 bf16
            int key = chunk >> 4, c = (chunk & 15) * 8;
            async_load_b128(&Ks[key][c], Kb,
                            (uint32_t)((tileBase + (size_t)key * HD + c) * 2));
            v8bf vv = *(const v8bf*)(Vg + (size_t)key * HD + c);
#pragma unroll
            for (int j = 0; j < 8; ++j) VTs[c + j][key] = vv[j];
        }
        if (kt < ktEnd) {
            // prefetch next tile, one 128B cacheline per thread per matrix
            __builtin_prefetch(Kb + tileBase + (size_t)128 * HD + (size_t)tid * 64, 0, 1);
            __builtin_prefetch(Vg + (size_t)128 * HD + (size_t)tid * 64, 0, 1);
        }
        wait_async0();
        __syncthreads();

        // S = Q * K^T  (8 key fragments of 16)
        v8f st[8];
#pragma unroll
        for (int n = 0; n < 8; ++n) {
            v8f s = zero;
#pragma unroll
            for (int kd = 0; kd < 4; ++kd) {
                v16bf kf = frag_b(&Ks[n * 16 + lr][kd * 32 + half * 16]);
                s = __builtin_amdgcn_wmma_f32_16x16x32_bf16(
                    false, qf[kd], false, kf, (short)0, s, false, false);
            }
            st[n] = s;
        }
        // scale + causal mask (lane holds column kg, rows half*8+r)
#pragma unroll
        for (int n = 0; n < 8; ++n) {
            int kg = k0 + n * 16 + lr;
#pragma unroll
            for (int r = 0; r < 8; ++r) {
                int qg = q0 + w * 16 + half * 8 + r;
                float x = st[n][r] * scale;
                st[n][r] = (kg > qg) ? -3.0e38f : x;
            }
        }
        // streaming softmax update
        float alpha[8], mnew[8], rs[8];
#pragma unroll
        for (int r = 0; r < 8; ++r) {
            float mx = -3.0e38f;
#pragma unroll
            for (int n = 0; n < 8; ++n) mx = fmaxf(mx, st[n][r]);
            mx = redmax16(mx);
            mnew[r] = fmaxf(mrow[r], mx);
            alpha[r] = __expf(mrow[r] - mnew[r]);
            mrow[r] = mnew[r];
            rs[r] = 0.f;
        }
#pragma unroll
        for (int n = 0; n < 8; ++n)
#pragma unroll
            for (int r = 0; r < 8; ++r) {
                float p = __expf(st[n][r] - mnew[r]);
                st[n][r] = p;
                rs[r] += p;
            }
#pragma unroll
        for (int r = 0; r < 8; ++r) {
            rs[r] = redsum16(rs[r]);
            lrow[r] = lrow[r] * alpha[r] + rs[r];
        }
#pragma unroll
        for (int nd = 0; nd < 8; ++nd)
#pragma unroll
            for (int r = 0; r < 8; ++r) o[nd][r] *= alpha[r];

        // P: C-layout -> per-wave LDS -> A-layout fragments (wave-local, DS in-order)
#pragma unroll
        for (int n = 0; n < 8; ++n)
#pragma unroll
            for (int r = 0; r < 8; ++r)
                Ps[w][half * 8 + r][n * 16 + lr] = (__bf16)st[n][r];
        v16bf pf[4];
#pragma unroll
        for (int kc = 0; kc < 4; ++kc)
            pf[kc] = frag_a(&Ps[w][lr][kc * 32 + half * 16]);
        // O += P * V
#pragma unroll
        for (int nd = 0; nd < 8; ++nd)
#pragma unroll
            for (int kc = 0; kc < 4; ++kc) {
                v16bf vf = frag_b(&VTs[nd * 16 + lr][kc * 32 + half * 16]);
                o[nd] = __builtin_amdgcn_wmma_f32_16x16x32_bf16(
                    false, pf[kc], false, vf, (short)0, o[nd], false, false);
            }
    }
    // normalize and store attn rows (layout (s*B+b, h*128+d)) as bf16
#pragma unroll
    for (int nd = 0; nd < 8; ++nd)
#pragma unroll
        for (int r = 0; r < 8; ++r) {
            int sg = q0 + w * 16 + half * 8 + r;
            int d  = nd * 16 + lr;
            float val = o[nd][r] / lrow[r];
            attn[((size_t)(sg * 2 + b)) * 2048 + h * 128 + d] = (__bf16)val;
        }
}

extern "C" void kernel_launch(void* const* d_in, const int* in_sizes, int n_in,
                              void* d_out, int out_size, void* d_ws, size_t ws_size,
                              hipStream_t stream) {
    (void)in_sizes; (void)n_in; (void)out_size; (void)ws_size;
    const float* hidden = (const float*)d_in[0];   // (2048,2,2048)
    const float* wqkv   = (const float*)d_in[1];   // (6144,2048)
    const float* wout   = (const float*)d_in[2];   // (2048,2048)

    char* ws = (char*)d_ws;
    size_t off = 0;
    __bf16* Xbf    = (__bf16*)(ws + off); off += (size_t)4096 * 2048 * 2;
    __bf16* Wqkvbf = (__bf16*)(ws + off); off += (size_t)6144 * 2048 * 2;
    __bf16* Woutbf = (__bf16*)(ws + off); off += (size_t)2048 * 2048 * 2;
    float*  QKVf   = (float*) (ws + off); off += (size_t)4096 * 6144 * 4;
    __bf16* Qbf    = (__bf16*)(ws + off); off += (size_t)32 * 2048 * 128 * 2;
    __bf16* Kbf    = (__bf16*)(ws + off); off += (size_t)32 * 2048 * 128 * 2;
    __bf16* Vbf    = (__bf16*)(ws + off); off += (size_t)32 * 2048 * 128 * 2;
    __bf16* attnbf = (__bf16*)(ws + off); off += (size_t)4096 * 2048 * 2;

    // 1) convert inputs to bf16
    {
        long n = (long)4096 * 2048;
        f32_to_bf16_kernel<<<(unsigned)((n + 255) / 256), 256, 0, stream>>>(hidden, Xbf, n);
    }
    {
        long n = (long)6144 * 2048;
        f32_to_bf16_kernel<<<(unsigned)((n + 255) / 256), 256, 0, stream>>>(wqkv, Wqkvbf, n);
    }
    {
        long n = (long)2048 * 2048;
        f32_to_bf16_kernel<<<(unsigned)((n + 255) / 256), 256, 0, stream>>>(wout, Woutbf, n);
    }
    // 2) QKV projection: (4096 x 2048) * (6144 x 2048)^T -> f32
    gemm_bf16_kernel<<<dim3(6144 / 128, 4096 / 128), 256, 0, stream>>>(
        Xbf, Wqkvbf, QKVf, 4096, 6144, 2048);
    // 3) RoPE + split into (b,h,s,d) bf16
    {
        long n = (long)4096 * 6144;
        rope_split_kernel<<<(unsigned)((n + 255) / 256), 256, 0, stream>>>(QKVf, Qbf, Kbf, Vbf);
    }
    // 4) causal flash attention
    flash_kernel<<<dim3(2048 / 128, 32), 256, 0, stream>>>(Qbf, Kbf, Vbf, attnbf);
    // 5) output projection: (4096 x 2048) * (2048 x 2048)^T -> d_out (f32)
    gemm_bf16_kernel<<<dim3(2048 / 128, 4096 / 128), 256, 0, stream>>>(
        attnbf, Woutbf, (float*)d_out, 4096, 2048, 2048);
}
